// HybridGATModel_47906065219693
// MI455X (gfx1250) — matrix operand order repaired and checked by reading, not compiled, not run on hardware
//
#include <hip/hip_runtime.h>

// ---------------------------------------------------------------------------
// HybridGATModel forward for MI455X (gfx1250).
// Heavy GEMMs: v_wmma_f32_16x16x32_bf16 (bf16 in, f32 acc). Panels are staged
// into LDS in *fragment order* so per-lane operand loads are two ds_load_b128
// instead of 16 ds_load_u16. Each wave computes a 16x32 tile (2 WMMAs sharing
// one A fragment). global_prefetch_b8 prefetches the next K panel branchlessly.
// Whole working set (~50MB) is L2-resident; HBM is irrelevant.
// ---------------------------------------------------------------------------

typedef __attribute__((ext_vector_type(16))) __bf16 v16bf;
typedef __attribute__((ext_vector_type(8)))  float  v8f;

__device__ __forceinline__ unsigned int f2bf(float f) {
  unsigned int u = __float_as_uint(f);
  unsigned int r = u + 0x7FFFu + ((u >> 16) & 1u);   // round-to-nearest-even
  return r >> 16;
}

// ------------------------------ conv1 (3->64) ------------------------------
__global__ void conv1_bn_relu(const float* __restrict__ img, const float* __restrict__ w,
                              const float* __restrict__ b,  const float* __restrict__ g,
                              const float* __restrict__ bb, const float* __restrict__ m,
                              const float* __restrict__ v,  float* __restrict__ out) {
  int idx = blockIdx.x * 256 + threadIdx.x;       // 8*64*64*64 = 2097152
  int x  = idx & 63;
  int y  = (idx >> 6) & 63;
  int co = (idx >> 12) & 63;
  int bI = idx >> 18;
  float acc = 0.f;
  const float* wp = w + co * 27;
  for (int ci = 0; ci < 3; ++ci) {
    for (int ky = 0; ky < 3; ++ky) {
      int iy = y + ky - 1; if (iy < 0 || iy > 63) continue;
      for (int kx = 0; kx < 3; ++kx) {
        int ix = x + kx - 1; if (ix < 0 || ix > 63) continue;
        acc += img[((bI * 3 + ci) * 64 + iy) * 64 + ix] * wp[(ci * 3 + ky) * 3 + kx];
      }
    }
  }
  acc += b[co];
  float inv = g[co] * rsqrtf(v[co] + 1e-5f);
  acc = acc * inv + (bb[co] - m[co] * inv);
  out[idx] = fmaxf(acc, 0.f);
}

// ------------------------------ 2x2 maxpool --------------------------------
__global__ void maxpool2(const float* __restrict__ in, float* __restrict__ out) {
  int idx = blockIdx.x * 256 + threadIdx.x;       // 8*64*32*32 = 524288
  int x  = idx & 31;
  int y  = (idx >> 5) & 31;
  int c  = (idx >> 10) & 63;
  int bI = idx >> 16;
  const float* p = in + ((size_t)(bI * 64 + c) * 64 + y * 2) * 64 + x * 2;
  out[idx] = fmaxf(fmaxf(p[0], p[1]), fmaxf(p[64], p[65]));
}

// --------------------- conv2 (64->128), node-major out ---------------------
__global__ void conv2_bn_relu(const float* __restrict__ in, const float* __restrict__ w,
                              const float* __restrict__ b,  const float* __restrict__ g,
                              const float* __restrict__ bb, const float* __restrict__ m,
                              const float* __restrict__ v,  float* __restrict__ X) {
  int idx = blockIdx.x * 256 + threadIdx.x;       // 8*128*32*32 = 1048576
  int x  = idx & 31;
  int y  = (idx >> 5) & 31;
  int co = (idx >> 10) & 127;
  int bI = idx >> 17;
  float acc = 0.f;
  const float* wp = w + (size_t)co * 64 * 9;
  for (int ci = 0; ci < 64; ++ci) {
    const float* ip = in + (size_t)(bI * 64 + ci) * 1024;
    const float* wc = wp + ci * 9;
    for (int ky = 0; ky < 3; ++ky) {
      int iy = y + ky - 1; if (iy < 0 || iy > 31) continue;
      for (int kx = 0; kx < 3; ++kx) {
        int ix = x + kx - 1; if (ix < 0 || ix > 31) continue;
        acc += ip[iy * 32 + ix] * wc[ky * 3 + kx];
      }
    }
  }
  acc += b[co];
  float inv = g[co] * rsqrtf(v[co] + 1e-5f);
  acc = acc * inv + (bb[co] - m[co] * inv);
  int node = bI * 1024 + y * 32 + x;              // node-major for the GAT GEMMs
  X[(size_t)node * 128 + co] = fmaxf(acc, 0.f);
}

// --------------------- bf16 WMMA GEMM: C = A(f32) * B(f32) -----------------
// Block tile 64(M) x 64(N), K-step 32; 8 waves: wave (mt,nt) owns a 16x32
// output tile = 2 accumulators sharing one A fragment.
// LDS panels are stored in WMMA *fragment order*:
//   A: [mtile 0..3][lane 0..31][16 bf16]   (lane's v16bf is 32B contiguous)
//   B: [ntile 0..3][lane 0..31][16 bf16]
// Fragment mapping (CDNA5 16-bit layouts):
//   A elem (m,k): lane = ((k>>3)&1)*16 + m, ushort = ((k>>4)<<3) + (k&7)
//   B elem (k,n): lane = ((k>>3)&1)*16 + n, ushort = ((k>>4)<<3) + (k&7)
// All call sites have M%64==0, N%64==0, K%32==0 -> no edge guards.
#define BM 64
#define BN 64
#define BK 32

__global__ __launch_bounds__(256) void gemm_bf16_wmma(
    const float* __restrict__ A, const float* __restrict__ B, float* __restrict__ C,
    int M, int N, int K, int lda, int ldb, int ldc) {
  __shared__ __attribute__((aligned(32))) unsigned int Au[4 * 32 * 8];  // 4KB
  __shared__ __attribute__((aligned(32))) unsigned int Bu[4 * 32 * 8];  // 4KB

  const int t      = threadIdx.x;
  const int wave   = t >> 5;
  const int lane   = t & 31;
  const int half   = lane >> 4;
  const int lm     = lane & 15;
  const int mt     = wave >> 1;      // 0..3  (16 rows each)
  const int nt     = wave & 1;       // 0..1  (32 cols each)
  const int mBlock = blockIdx.y * BM;
  const int nBlock = blockIdx.x * BN;

  // Staging coordinates (computed once, reused each K step).
  // A: 1024 bf16-pairs (64 rows x 16 even-col pairs), 4 per thread.
  // B: 1024 bf16-pairs (16 even-k pairs x 64 cols), 4 per thread.
  v8f acc0 = {0.f, 0.f, 0.f, 0.f, 0.f, 0.f, 0.f, 0.f};
  v8f acc1 = {0.f, 0.f, 0.f, 0.f, 0.f, 0.f, 0.f, 0.f};

  for (int kk = 0; kk < K; kk += BK) {
    const int kpf = (kk + BK < K) ? BK : 0;   // branchless prefetch distance

#pragma unroll
    for (int i = 0; i < 4; ++i) {
      int idx = t + i * 256;                  // 0..1023
      int r   = idx >> 4;                     // 0..63
      int cp  = (idx & 15) << 1;              // even col 0..30
      const float* ap = &A[(size_t)(mBlock + r) * lda + kk + cp];
      unsigned int pk = f2bf(ap[0]) | (f2bf(ap[1]) << 16);
      __builtin_prefetch(ap + kpf, 0, 1);     // global_prefetch_b8
      int tile = r >> 4, mm = r & 15;
      int ln   = (((cp >> 3) & 1) << 4) + mm;
      int u    = ((cp >> 4) << 3) + (cp & 7); // even
      Au[(tile * 32 + ln) * 8 + (u >> 1)] = pk;
    }
#pragma unroll
    for (int i = 0; i < 4; ++i) {
      int idx = t + i * 256;                  // 0..1023
      int kp  = (idx >> 6) << 1;              // even k 0..30
      int n   = idx & 63;                     // 0..63
      const float* bp = &B[(size_t)(kk + kp) * ldb + nBlock + n];
      unsigned int pk = f2bf(bp[0]) | (f2bf(bp[ldb]) << 16);
      __builtin_prefetch(bp + (size_t)kpf * ldb, 0, 1);
      int tile = n >> 4, nn = n & 15;
      int ln   = (((kp >> 3) & 1) << 4) + nn;
      Bu[(tile * 32 + ln) * 8 + (((kp >> 4) << 3) + (kp & 7)) / 2] = pk;
    }
    __syncthreads();

    v16bf fa  = *(const v16bf*)(Au + (mt * 32 + lane) * 8);
    v16bf fb0 = *(const v16bf*)(Bu + ((nt * 2 + 0) * 32 + lane) * 8);
    v16bf fb1 = *(const v16bf*)(Bu + ((nt * 2 + 1) * 32 + lane) * 8);

    acc0 = __builtin_amdgcn_wmma_f32_16x16x32_bf16(false, fa, false, fb0,
                                                   (short)0, acc0, false, false);
    acc1 = __builtin_amdgcn_wmma_f32_16x16x32_bf16(false, fa, false, fb1,
                                                   (short)0, acc1, false, false);
    __syncthreads();
  }

  // C/D layout: lanes 0-15 -> rows r, lanes 16-31 -> rows r+8; col = lane&15.
  int gm  = mBlock + mt * 16 + half * 8;
  int gn0 = nBlock + nt * 32 + lm;
#pragma unroll
  for (int r = 0; r < 8; ++r) {
    C[(size_t)(gm + r) * ldc + gn0]      = acc0[r];
    C[(size_t)(gm + r) * ldc + gn0 + 16] = acc1[r];
  }
}

// --------------------- per-node attention scores (s, d) --------------------
__global__ void gat_scores(const float* __restrict__ H, const float* __restrict__ asrc,
                           const float* __restrict__ adst, float* __restrict__ s,
                           float* __restrict__ d, int N, int D) {
  int i = blockIdx.x * blockDim.x + threadIdx.x;
  if (i >= N) return;
  const float* h = H + (size_t)i * D;
  float ss = 0.f, dd = 0.f;
  for (int f = 0; f < D; ++f) { float v = h[f]; ss += v * asrc[f]; dd += v * adst[f]; }
  s[i] = ss; d[i] = dd;
}

// --------- dense softmax row per dst node i in [0,1024): alpha[i][j] -------
__global__ void gat_alpha(const float* __restrict__ s, const float* __restrict__ d,
                          float* __restrict__ alpha) {
  const int nN = 1024;
  int i = blockIdx.x;
  int t = threadIdx.x;
  float di = d[i];
  __shared__ float red[256];
  float m = -1e30f;
  for (int j = t; j < nN; j += 256) {
    float e = s[j] + di; e = e > 0.f ? e : 0.2f * e;
    m = fmaxf(m, e);
  }
  red[t] = m; __syncthreads();
  for (int off = 128; off > 0; off >>= 1) {
    if (t < off) red[t] = fmaxf(red[t], red[t + off]);
    __syncthreads();
  }
  m = red[0]; __syncthreads();
  float sum = 0.f;
  for (int j = t; j < nN; j += 256) {
    float e = s[j] + di; e = e > 0.f ? e : 0.2f * e;
    sum += __expf(e - m);
  }
  red[t] = sum; __syncthreads();
  for (int off = 128; off > 0; off >>= 1) {
    if (t < off) red[t] += red[t + off];
    __syncthreads();
  }
  float inv = 1.0f / red[0];
  for (int j = t; j < nN; j += 256) {
    float e = s[j] + di; e = e > 0.f ? e : 0.2f * e;
    alpha[(size_t)i * nN + j] = __expf(e - m) * inv;
  }
}

// --------- epilogue: bias + relu; rows >=1024 are pure self-loops ----------
__global__ void gat_epilogue(const float* __restrict__ H, const float* __restrict__ bias,
                             float* __restrict__ Y, int dShift) {
  int idx = blockIdx.x * 256 + threadIdx.x;       // 8192 << dShift total
  int f = idx & ((1 << dShift) - 1);
  int i = idx >> dShift;
  float val = ((i < 1024) ? Y[idx] : H[idx]) + bias[f];
  Y[idx] = fmaxf(val, 0.f);
}

// ------------------------------ mean pool ----------------------------------
__global__ void mean_pool(const float* __restrict__ Y, float* __restrict__ pooled) {
  int idx = blockIdx.x * 256 + threadIdx.x;       // 8*128 = 1024
  int f  = idx & 127;
  int bI = idx >> 7;
  const float* p = Y + (size_t)bI * 1024 * 128 + f;
  float s = 0.f;
  for (int n = 0; n < 1024; ++n) s += p[(size_t)n * 128];
  pooled[idx] = s * (1.0f / 1024.0f);
}

// --------------------- classifier + log_softmax (8x10) ---------------------
__global__ void classifier(const float* __restrict__ pooled, const float* __restrict__ W,
                           const float* __restrict__ b, float* __restrict__ out) {
  __shared__ float ll[8][10];
  int t = threadIdx.x;
  if (t < 80) {
    int bI = t / 10, j = t % 10;
    float acc = b[j];
    for (int f = 0; f < 128; ++f) acc += pooled[bI * 128 + f] * W[f * 10 + j];
    ll[bI][j] = acc;
  }
  __syncthreads();
  if (t < 80) {
    int bI = t / 10, j = t % 10;
    float m = ll[bI][0];
    for (int q = 1; q < 10; ++q) m = fmaxf(m, ll[bI][q]);
    float sum = 0.f;
    for (int q = 0; q < 10; ++q) sum += __expf(ll[bI][q] - m);
    out[t] = (ll[bI][j] - m) - __logf(sum);
  }
}

// ---------------------------------------------------------------------------
extern "C" void kernel_launch(void* const* d_in, const int* in_sizes, int n_in,
                              void* d_out, int out_size, void* d_ws, size_t ws_size,
                              hipStream_t stream) {
  (void)in_sizes; (void)n_in; (void)out_size; (void)ws_size;
  const float* images   = (const float*)d_in[0];
  const float* conv1_w  = (const float*)d_in[1];
  const float* conv1_b  = (const float*)d_in[2];
  const float* bn1_g    = (const float*)d_in[3];
  const float* bn1_b    = (const float*)d_in[4];
  const float* bn1_m    = (const float*)d_in[5];
  const float* bn1_v    = (const float*)d_in[6];
  const float* conv2_w  = (const float*)d_in[7];
  const float* conv2_b  = (const float*)d_in[8];
  const float* bn2_g    = (const float*)d_in[9];
  const float* bn2_b    = (const float*)d_in[10];
  const float* bn2_m    = (const float*)d_in[11];
  const float* bn2_v    = (const float*)d_in[12];
  const float* gat1_w   = (const float*)d_in[13];
  const float* gat1_as  = (const float*)d_in[14];
  const float* gat1_ad  = (const float*)d_in[15];
  const float* gat1_bi  = (const float*)d_in[16];
  const float* gat2_w   = (const float*)d_in[17];
  const float* gat2_as  = (const float*)d_in[18];
  const float* gat2_ad  = (const float*)d_in[19];
  const float* gat2_bi  = (const float*)d_in[20];
  const float* out_w    = (const float*)d_in[21];
  const float* out_b    = (const float*)d_in[22];

  float* ws = (float*)d_ws;
  size_t o = 0;
  float* x1  = ws + o; o += (size_t)8 * 64 * 64 * 64;   // conv1 out
  float* x2  = ws + o; o += (size_t)8 * 64 * 32 * 32;   // pooled
  float* X   = ws + o; o += (size_t)8192 * 128;         // node features
  float* H1  = ws + o; o += (size_t)8192 * 256;
  float* s1  = ws + o; o += 8192;
  float* d1  = ws + o; o += 8192;
  float* al1 = ws + o; o += (size_t)1024 * 1024;
  float* Y1  = ws + o; o += (size_t)8192 * 256;
  float* H2  = ws + o; o += (size_t)8192 * 128;
  float* s2  = ws + o; o += 8192;
  float* d2  = ws + o; o += 8192;
  float* al2 = ws + o; o += (size_t)1024 * 1024;
  float* Y2  = ws + o; o += (size_t)8192 * 128;
  float* pooled = ws + o; o += 1024;

  // CNN front-end
  conv1_bn_relu<<<8192, 256, 0, stream>>>(images, conv1_w, conv1_b,
                                          bn1_g, bn1_b, bn1_m, bn1_v, x1);
  maxpool2<<<2048, 256, 0, stream>>>(x1, x2);
  conv2_bn_relu<<<4096, 256, 0, stream>>>(x2, conv2_w, conv2_b,
                                          bn2_g, bn2_b, bn2_m, bn2_v, X);

  // GAT layer 1: H1 = X @ W1   (8192 x 256, K=128)
  {
    dim3 g(256 / BN, 8192 / BM);
    gemm_bf16_wmma<<<g, 256, 0, stream>>>(X, gat1_w, H1, 8192, 256, 128, 128, 256, 256);
  }
  gat_scores<<<32, 256, 0, stream>>>(H1, gat1_as, gat1_ad, s1, d1, 8192, 256);
  gat_alpha<<<1024, 256, 0, stream>>>(s1, d1, al1);
  {  // Y1[0:1024] = alpha1 @ H1[0:1024]   (1024 x 256, K=1024)
    dim3 g(256 / BN, 1024 / BM);
    gemm_bf16_wmma<<<g, 256, 0, stream>>>(al1, H1, Y1, 1024, 256, 1024, 1024, 256, 256);
  }
  gat_epilogue<<<8192, 256, 0, stream>>>(H1, gat1_bi, Y1, 8);

  // GAT layer 2: H2 = Y1 @ W2   (8192 x 128, K=256)
  {
    dim3 g(128 / BN, 8192 / BM);
    gemm_bf16_wmma<<<g, 256, 0, stream>>>(Y1, gat2_w, H2, 8192, 128, 256, 256, 128, 128);
  }
  gat_scores<<<32, 256, 0, stream>>>(H2, gat2_as, gat2_ad, s2, d2, 8192, 128);
  gat_alpha<<<1024, 256, 0, stream>>>(s2, d2, al2);
  {  // Y2[0:1024] = alpha2 @ H2[0:1024]   (1024 x 128, K=1024)
    dim3 g(128 / BN, 1024 / BM);
    gemm_bf16_wmma<<<g, 256, 0, stream>>>(al2, H2, Y2, 1024, 128, 1024, 1024, 128, 128);
  }
  gat_epilogue<<<4096, 256, 0, stream>>>(H2, gat2_bi, Y2, 7);

  // Head
  mean_pool<<<4, 256, 0, stream>>>(Y2, pooled);
  classifier<<<1, 96, 0, stream>>>(pooled, out_w, out_b, (float*)d_out);
}
